// AttentionLayer_11682311045645
// MI455X (gfx1250) — matrix operand order repaired and checked by reading, not compiled, not run on hardware
//
#include <hip/hip_runtime.h>

// AttentionLayer for MI455X (gfx1250): f16 WMMA pipeline, flash-attention
// streaming over keys (transposed-score formulation keeps softmax per-lane).
// K/V chunks staged into LDS once per block via gfx1250 async global->LDS
// DMA (ASYNCcnt), double-buffered against the WMMA work.

#define BB 8
#define SS 2048
#define DD 256
#define UU 128

typedef __attribute__((ext_vector_type(16))) _Float16 v16h;
typedef __attribute__((ext_vector_type(8)))  _Float16 v8h;
typedef __attribute__((ext_vector_type(8)))  float    v8f;
typedef __attribute__((ext_vector_type(4)))  int      v4i;

#if defined(__has_builtin)
#if __has_builtin(__builtin_amdgcn_global_load_async_to_lds_b128)
#define HAVE_ASYNC_LDS 1
#endif
#if __has_builtin(__builtin_amdgcn_s_wait_asynccnt)
#define HAVE_WAIT_ASYNC 1
#endif
#endif

#if defined(HAVE_WAIT_ASYNC)
#define WAIT_ASYNC(n) __builtin_amdgcn_s_wait_asynccnt((unsigned short)(n))
#else
#define WAIT_ASYNC(n) asm volatile("s_wait_asynccnt %0" ::"n"(n) : "memory")
#endif

static __device__ __forceinline__ v8f wmma32(v16h a, v16h b, v8f c) {
  // D = A(16x32 f16) * B(32x16 f16) + C(16x16 f32)
  return __builtin_amdgcn_wmma_f32_16x16x32_f16(false, a, false, b, (short)0, c,
                                                false, false);
}

// A-fragment (16x32, row-major source, K contiguous within a row).
// Contiguous-K layout used here: lane m=lane&15; h=lane>>4; the 16 halves are K=16h..16h+15 in order.
static __device__ __forceinline__ v16h frag_a_f16(const _Float16* tile, int ld) {
  const int lane = threadIdx.x & 31;
  const int m = lane & 15, h = lane >> 4;
  const _Float16* p = tile + (size_t)m * ld;
  v8h lo = *(const v8h*)(p + 16 * h);
  v8h hi = *(const v8h*)(p + 16 * h + 8);
  v16h r;
#pragma unroll
  for (int i = 0; i < 8; ++i) { r[i] = lo[i]; r[i + 8] = hi[i]; }
  return r;
}

// Same A-fragment but converting from an f32 row-major source on the fly.
static __device__ __forceinline__ v16h frag_a_f32(const float* tile, int ld) {
  const int lane = threadIdx.x & 31;
  const int m = lane & 15, h = lane >> 4;
  const float* p = tile + (size_t)m * ld;
  v16h r;
#pragma unroll
  for (int i = 0; i < 8; ++i) r[i] = (_Float16)p[16 * h + i];
#pragma unroll
  for (int i = 0; i < 8; ++i) r[i + 8] = (_Float16)p[16 * h + 8 + i];
  return r;
}

// B-fragment (32x16). Source stored [n][k] with K contiguous.
// Contiguous-K layout used here: lane n=lane&15; h=lane>>4; the 16 halves are K=16h..16h+15 in order.
static __device__ __forceinline__ v16h frag_b_rowK(const _Float16* tile, int ld) {
  const int lane = threadIdx.x & 31;
  const int n = lane & 15, h = lane >> 4;
  const _Float16* p = tile + (size_t)n * ld + 16 * h;
  v8h lo = *(const v8h*)p;
  v8h hi = *(const v8h*)(p + 8);
  v16h r;
#pragma unroll
  for (int i = 0; i < 8; ++i) { r[i] = lo[i]; r[i + 8] = hi[i]; }
  return r;
}

static __device__ __forceinline__ unsigned pack2h(float a, float b) {
  union { _Float16 h[2]; unsigned u; } t;
  t.h[0] = (_Float16)a; t.h[1] = (_Float16)b;
  return t.u;
}

// 16-byte global -> LDS copy: async DMA on gfx1250 (tracked by ASYNCcnt).
// Builtin signature takes int4* pointers (AS1 source, AS3 destination).
static __device__ __forceinline__ void cp_g2l_b128(const _Float16* g,
                                                   _Float16* l) {
#if defined(HAVE_ASYNC_LDS)
  __builtin_amdgcn_global_load_async_to_lds_b128(
      (__attribute__((address_space(1))) v4i*)g,
      (__attribute__((address_space(3))) v4i*)l, 0, 0);
#else
  *(v8h*)l = *(const v8h*)g;
#endif
}

// ---------------------------------------------------------------------------
// Kernel 1: QKV projection.  X[BS x D](f32) @ W[D x U](f32) -> f16.
// blockIdx.y selects Q/K/V; W is transposed into LDS as f16 so B-frags are
// contiguous-K ds loads.  Q,K stored row-major [BS][U]; V stored transposed
// [B][U][S] for the PV GEMM later.
// ---------------------------------------------------------------------------
#define WT_PITCH 272  // halves: 256 + 16 pad (16B-aligned rows, conflict pad)

__global__ __launch_bounds__(256) void qkv_proj_kernel(
    const float* __restrict__ X, const float* __restrict__ Wq,
    const float* __restrict__ Wk, const float* __restrict__ Wv,
    _Float16* __restrict__ Qh, _Float16* __restrict__ Kh,
    _Float16* __restrict__ Vt) {
  __shared__ _Float16 ldsWt[UU * WT_PITCH];  // [u][d] f16, ~69.6 KB
  const int which = blockIdx.y;
  const float* W = (which == 0) ? Wq : (which == 1) ? Wk : Wv;
  for (int idx = threadIdx.x; idx < DD * UU; idx += blockDim.x) {
    const int d = idx >> 7;         // W is [D][U], U=128
    const int u = idx & (UU - 1);
    ldsWt[u * WT_PITCH + d] = (_Float16)W[idx];
  }
  __syncthreads();

  const int wave = threadIdx.x >> 5;
  const int lane = threadIdx.x & 31;
  const int m0 = blockIdx.x * 128 + wave * 16;  // global row tile (16 rows)

  v8f acc[8] = {};
#pragma unroll
  for (int kc = 0; kc < DD / 32; ++kc) {
    v16h a = frag_a_f32(X + (size_t)m0 * DD + kc * 32, DD);
#pragma unroll
    for (int nt = 0; nt < 8; ++nt) {
      v16h b = frag_b_rowK(&ldsWt[(nt * 16) * WT_PITCH + kc * 32], WT_PITCH);
      acc[nt] = wmma32(a, b, acc[nt]);
    }
  }

  const int h = lane >> 4, ln = lane & 15;
  if (which < 2) {
    _Float16* dst = (which == 0) ? Qh : Kh;
#pragma unroll
    for (int nt = 0; nt < 8; ++nt)
#pragma unroll
      for (int i = 0; i < 8; ++i)
        dst[(size_t)(m0 + i + 8 * h) * UU + nt * 16 + ln] = (_Float16)acc[nt][i];
  } else {
    const int b = m0 / SS, s0 = m0 % SS;  // 128-row tiles never straddle a batch
#pragma unroll
    for (int nt = 0; nt < 8; ++nt) {
      const int u = nt * 16 + ln;
      v8h pk;
#pragma unroll
      for (int i = 0; i < 8; ++i) pk[i] = (_Float16)acc[nt][i];
      *(v8h*)&Vt[((size_t)b * UU + u) * SS + s0 + 8 * h] = pk;  // b128 store
    }
  }
}

// ---------------------------------------------------------------------------
// Kernel 2: flash attention.  4 waves/block share one batch; each wave owns a
// 16-query tile.  Per 32-key chunk, the block stages K (32x128) and V^T
// (128x32) into double-buffered LDS with async global->LDS DMA, then all
// waves build A-frags from LDS.  scores^T = K*Q^T keeps softmax per-lane;
// out^T accumulated as V^T * P^T.
// ---------------------------------------------------------------------------
#define KP 136      // ldsK row pitch (halves): 128 + 8 -> conflict-free b128
#define VP 40       // ldsV row pitch (halves): 32 + 8  -> conflict-free b128
#define NCHUNK (SS / 32)

// Stage one 32-key chunk: K rows [kc*32, +32) and V^T columns [kc*32, +32).
// 8 b128 copies per wave (4 K + 4 V) -> ASYNCcnt grows by 8 per chunk.
static __device__ __forceinline__ void stage_chunk(
    const _Float16* __restrict__ Kb, const _Float16* __restrict__ Vb, int kc,
    _Float16* ldsK, _Float16* ldsV) {
  const int t = threadIdx.x;  // 0..127
#pragma unroll
  for (int r = 0; r < 4; ++r) {
    const int ik = t + r * 128;            // 512 segments of K tile
    const int s = ik >> 4, seg = ik & 15;  // row s (32), 16B segment (16)
    cp_g2l_b128(Kb + (size_t)(kc * 32 + s) * UU + seg * 8,
                ldsK + s * KP + seg * 8);
  }
#pragma unroll
  for (int r = 0; r < 4; ++r) {
    const int iv = t + r * 128;            // 512 segments of V^T tile
    const int u = iv >> 2, seg = iv & 3;   // row u (128), 16B segment (4)
    cp_g2l_b128(Vb + (size_t)u * SS + kc * 32 + seg * 8,
                ldsV + u * VP + seg * 8);
  }
}

__global__ __launch_bounds__(128) void attn_kernel(
    const _Float16* __restrict__ Qh, const _Float16* __restrict__ Kh,
    const _Float16* __restrict__ Vt, _Float16* __restrict__ Oh) {
  __shared__ _Float16 ldsK[2][32 * KP];   // 17.0 KB
  __shared__ _Float16 ldsV[2][UU * VP];   // 20.0 KB

  const int wave = threadIdx.x >> 5;
  const int lane = threadIdx.x & 31;
  const int h = lane >> 4;
  const int qt = blockIdx.x * 4 + wave;  // 0..1023 query tiles
  const int b = qt >> 7;                 // SS/16 = 128 tiles per batch
  const int q0 = (qt & 127) * 16;

  const _Float16* Qbase = Qh + ((size_t)b * SS + q0) * UU;
  const _Float16* Kb = Kh + (size_t)b * SS * UU;
  const _Float16* Vb = Vt + (size_t)b * UU * SS;

  const float f = 0.08838834764831845f * 1.4426950408889634f;  // scale*log2(e)

  v16h bq[4];  // Q^T B-frags, resident across the whole key loop
#pragma unroll
  for (int c = 0; c < 4; ++c) bq[c] = frag_b_rowK(Qbase + c * 32, UU);

  v8f o[8] = {};                  // out^T accum: lane=q, rows=u
  float mrun = -3.0e38f, lrun = 0.0f;

  stage_chunk(Kb, Vb, 0, ldsK[0], ldsV[0]);  // prologue: chunk 0 in flight

  for (int kc = 0; kc < NCHUNK; ++kc) {
    const int buf = kc & 1;
    if (kc + 1 < NCHUNK) {  // prefetch next chunk into the other buffer
      stage_chunk(Kb, Vb, kc + 1, ldsK[buf ^ 1], ldsV[buf ^ 1]);
      WAIT_ASYNC(8);        // the 8 older copies (current chunk) are done
    } else {
      WAIT_ASYNC(0);
    }
    __syncthreads();        // current buffer visible to all 4 waves

    const _Float16* Kl = ldsK[buf];
    const _Float16* Vl = ldsV[buf];

    v8f c0 = {}, c1 = {};   // scores^T: rows=key(0-15 / 16-31), cols=q
#pragma unroll
    for (int c = 0; c < 4; ++c) {
      v16h a0 = frag_a_f16(Kl + c * 32, KP);
      v16h a1 = frag_a_f16(Kl + 16 * KP + c * 32, KP);
      c0 = wmma32(a0, bq[c], c0);
      c1 = wmma32(a1, bq[c], c1);
    }
    // --- online softmax (per-lane: this lane owns query lane&15) ---
    float mloc = c0[0];
#pragma unroll
    for (int i = 1; i < 8; ++i) mloc = fmaxf(mloc, c0[i]);
#pragma unroll
    for (int i = 0; i < 8; ++i) mloc = fmaxf(mloc, c1[i]);
    mloc = fmaxf(mloc, __shfl_xor(mloc, 16, 32));  // fuse half-wave partials
    const float mnew = fmaxf(mrun, mloc);
    const float alpha = exp2f((mrun - mnew) * f);
    lrun *= alpha;
#pragma unroll
    for (int ut = 0; ut < 8; ++ut) o[ut] *= alpha;
    float p0[8], p1[8], ssum = 0.0f;
#pragma unroll
    for (int i = 0; i < 8; ++i) { p0[i] = exp2f((c0[i] - mnew) * f); ssum += p0[i]; }
#pragma unroll
    for (int i = 0; i < 8; ++i) { p1[i] = exp2f((c1[i] - mnew) * f); ssum += p1[i]; }
    lrun += ssum + __shfl_xor(ssum, 16, 32);
    mrun = mnew;
    // --- assemble P^T B-frag (keys 0..31 on the K axis, this lane's q) ---
    unsigned my0[4], my1[4], pr0[4], pr1[4];
#pragma unroll
    for (int g = 0; g < 4; ++g) {
      my0[g] = pack2h(p0[2 * g], p0[2 * g + 1]);
      my1[g] = pack2h(p1[2 * g], p1[2 * g + 1]);
    }
#pragma unroll
    for (int g = 0; g < 4; ++g) {
      pr0[g] = (unsigned)__shfl_xor((int)my0[g], 16, 32);
      pr1[g] = (unsigned)__shfl_xor((int)my1[g], 16, 32);
    }
    union { v16h v; unsigned u[8]; } bp;
#pragma unroll
    for (int g = 0; g < 4; ++g) {
      bp.u[g]     = h ? pr1[g] : my0[g];  // K = 16h + 2g, 2g+1
      bp.u[g + 4] = h ? my1[g] : pr0[g];  // K = 16h + 8 + 2g, 2g+1
    }
    // --- out^T += V^T * P^T ---
#pragma unroll
    for (int ut = 0; ut < 8; ++ut) {
      v16h av = frag_a_f16(Vl + ut * 16 * VP, VP);
      o[ut] = wmma32(av, bp.v, o[ut]);
    }
    __syncthreads();  // done reading buf before iter kc+1 overwrites it
  }

  const float inv = 1.0f / lrun;
  const int qn = lane & 15;
  _Float16* orow = Oh + ((size_t)b * SS + q0 + qn) * UU;
#pragma unroll
  for (int ut = 0; ut < 8; ++ut) {
    v8h pk;
#pragma unroll
    for (int i = 0; i < 8; ++i) pk[i] = (_Float16)(o[ut][i] * inv);
    *(v8h*)&orow[ut * 16 + 8 * h] = pk;  // contiguous u, b128 store
  }
}

// ---------------------------------------------------------------------------
// Kernel 3: output projection + bias + residual.
// attn[BS x U](f16) @ Wo[U x D] -> out[BS x D](f32), Wo^T staged in LDS f16.
// ---------------------------------------------------------------------------
#define WO_PITCH 136  // halves: 128 + 8 pad

__global__ __launch_bounds__(256) void out_proj_kernel(
    const _Float16* __restrict__ Oh, const float* __restrict__ Wo,
    const float* __restrict__ bo, const float* __restrict__ X,
    float* __restrict__ out) {
  __shared__ _Float16 ldsWt[DD * WO_PITCH];  // [d][u] f16, ~69.6 KB
  for (int idx = threadIdx.x; idx < UU * DD; idx += blockDim.x) {
    const int u = idx >> 8;  // Wo is [U][D], D=256
    const int d = idx & (DD - 1);
    ldsWt[d * WO_PITCH + u] = (_Float16)Wo[idx];
  }
  __syncthreads();

  const int wave = threadIdx.x >> 5;
  const int lane = threadIdx.x & 31;
  const int m0 = blockIdx.x * 128 + wave * 16;

  v8f acc[16] = {};
#pragma unroll
  for (int kc = 0; kc < UU / 32; ++kc) {
    v16h a = frag_a_f16(Oh + (size_t)m0 * UU + kc * 32, UU);
#pragma unroll
    for (int nt = 0; nt < 16; ++nt) {
      v16h b = frag_b_rowK(&ldsWt[(nt * 16) * WO_PITCH + kc * 32], WO_PITCH);
      acc[nt] = wmma32(a, b, acc[nt]);
    }
  }

  const int h = lane >> 4, ln = lane & 15;
#pragma unroll
  for (int nt = 0; nt < 16; ++nt) {
    const int d = nt * 16 + ln;
    const float bias = bo[d];
#pragma unroll
    for (int i = 0; i < 8; ++i) {
      const size_t row = (size_t)(m0 + i + 8 * h);
      out[row * DD + d] = acc[nt][i] + bias + X[row * DD + d];
    }
  }
}

// ---------------------------------------------------------------------------
extern "C" void kernel_launch(void* const* d_in, const int* in_sizes, int n_in,
                              void* d_out, int out_size, void* d_ws,
                              size_t ws_size, hipStream_t stream) {
  (void)in_sizes; (void)n_in; (void)out_size; (void)ws_size;
  const float* X  = (const float*)d_in[0];
  const float* Wq = (const float*)d_in[1];
  const float* Wk = (const float*)d_in[2];
  const float* Wv = (const float*)d_in[3];
  const float* Wo = (const float*)d_in[4];
  const float* bo = (const float*)d_in[5];
  float* out = (float*)d_out;

  const size_t nQ = (size_t)BB * SS * UU;  // 2M elements per f16 tensor
  _Float16* ws = (_Float16*)d_ws;          // needs 4*nQ*2 = 16.8 MB
  _Float16* Qh = ws;
  _Float16* Kh = ws + nQ;
  _Float16* Vt = ws + 2 * nQ;
  _Float16* Oh = ws + 3 * nQ;

  dim3 g1(BB * SS / 128, 3);
  qkv_proj_kernel<<<g1, dim3(256), 0, stream>>>(X, Wq, Wk, Wv, Qh, Kh, Vt);
  attn_kernel<<<dim3((BB * SS / 16) / 4), dim3(128), 0, stream>>>(Qh, Kh, Vt, Oh);
  out_proj_kernel<<<dim3(BB * SS / 128), dim3(256), 0, stream>>>(Oh, Wo, bo, X, out);
}
